// Lstm_50388556316832
// MI455X (gfx1250) — matrix operand order repaired and checked by reading, not compiled, not run on hardware
//
#include <hip/hip_runtime.h>
#include <hip/hip_bf16.h>

// Problem constants (match reference)
#define Bq   32
#define Tq   128
#define Vq   32000
#define Eq   512
#define Hq   512
#define Lq   3
#define EPSq 1e-5f

typedef __attribute__((ext_vector_type(16))) __bf16 v16bf;
typedef __attribute__((ext_vector_type(8)))  float  v8f;
typedef __attribute__((ext_vector_type(4)))  int    v4i;

union Frag { v16bf bf; v4i q[2]; };

__device__ __forceinline__ unsigned short f32_to_bf16(float f) {
    unsigned int u = __float_as_uint(f);
    unsigned int lsb = (u >> 16) & 1u;
    u += 0x7fffu + lsb;                 // round-to-nearest-even
    return (unsigned short)(u >> 16);
}

// fast sigmoid/tanh on the native v_exp_f32 path (serial LSTM cell is latency-bound)
__device__ __forceinline__ float sigmoidf_(float x) {
    return 1.0f / (1.0f + __expf(-x));
}
__device__ __forceinline__ float tanhf_(float x) {
    return 1.0f - 2.0f / (__expf(2.0f * x) + 1.0f);
}

// A fragment (16x32 bf16): lane half h, row r; v0-3 = K[h*8 .. +7], v4-7 = K[16+h*8 .. +7]
__device__ __forceinline__ v16bf load_a_frag(const unsigned short* row, int kk, int half) {
    Frag f;
    f.q[0] = *(const v4i*)(row + kk + half * 8);
    f.q[1] = *(const v4i*)(row + kk + 16 + half * 8);
    return f.bf;
}

// B fragment (32x16 bf16, K-contiguous storage): lane half h; v0-3 = K[h*8 .. +7], v4-7 = K[16+h*8 .. +7]
__device__ __forceinline__ v16bf load_b_frag(const unsigned short* col, int kk, int half) {
    Frag f;
    f.q[0] = *(const v4i*)(col + kk + half * 8);
    f.q[1] = *(const v4i*)(col + kk + 16 + half * 8);
    return f.bf;
}

// ---- gfx1250 async global->LDS copy (ASYNCcnt path) ----
// Probe-learned signature: (v4i addrspace(1)*, v4i addrspace(3)*, imm offset, imm cpol)
__device__ __forceinline__ void async_copy_b128(const unsigned short* src, unsigned short* dst) {
#if __has_builtin(__builtin_amdgcn_global_load_async_to_lds_b128)
    __builtin_amdgcn_global_load_async_to_lds_b128(
        (__attribute__((address_space(1))) v4i*)(void*)src,
        (__attribute__((address_space(3))) v4i*)dst, 0, 0);
#else
    unsigned ldsoff = (unsigned)(size_t)(__attribute__((address_space(3))) unsigned short*)dst;
    asm volatile("global_load_async_to_lds_b128 %0, %1, off"
                 :: "v"(ldsoff), "v"(src) : "memory");
#endif
}

#if __has_builtin(__builtin_amdgcn_s_wait_asynccnt)
#define WAIT_ASYNC(n) __builtin_amdgcn_s_wait_asynccnt(n)
#else
#define WAIT_ASYNC(n) asm volatile("s_wait_asynccnt %0" :: "i"(n) : "memory")
#endif

// ---------------- prep kernels ----------------

// W [L, K=H+E, N=4H] fp32 -> WT [L, N'=4H, K] bf16 with gate-interleaved columns:
// permuted column n' = 4*h + g  maps to original column  g*H + h.
// This makes every 16-col WMMA tile carry complete (f,i,g,o) quadruples.
__global__ void prep_w(const float* __restrict__ W, unsigned short* __restrict__ WT) {
    const long total = (long)Lq * (4 * Hq) * (Hq + Eq);
    long i = (long)blockIdx.x * blockDim.x + threadIdx.x;
    if (i >= total) return;
    int k = (int)(i % (Hq + Eq));
    long r = i / (Hq + Eq);
    int n = (int)(r % (4 * Hq));     // permuted column index
    int l = (int)(r / (4 * Hq));
    int g = n & 3, h = n >> 2;
    WT[i] = f32_to_bf16(W[((long)l * (Hq + Eq) + k) * (4 * Hq) + g * Hq + h]);
}

// Wh [H, V] fp32 -> WhT [V, H] bf16
__global__ void prep_wh(const float* __restrict__ Wh, unsigned short* __restrict__ WhT) {
    const long total = (long)Vq * Hq;
    long i = (long)blockIdx.x * blockDim.x + threadIdx.x;
    if (i >= total) return;
    int h = (int)(i % Hq);
    int v = (int)(i / Hq);
    WhT[i] = f32_to_bf16(Wh[(long)h * Vq + v]);
}

// x[b,t,e] = bf16(emb[idx[b,t], e])
__global__ void embed_gather(const int* __restrict__ idx, const float* __restrict__ emb,
                             unsigned short* __restrict__ x) {
    const long total = (long)Bq * Tq * Eq;
    long i = (long)blockIdx.x * blockDim.x + threadIdx.x;
    if (i >= total) return;
    int e = (int)(i % Eq);
    long bt = i / Eq;
    int tok = idx[bt];
    x[i] = f32_to_bf16(emb[(long)tok * Eq + e]);
}

// zero h(bf16) and c(f32): [L,B,H]
__global__ void init_state(unsigned short* __restrict__ hbf, float* __restrict__ c) {
    int i = blockIdx.x * blockDim.x + threadIdx.x;
    if (i >= Lq * Bq * Hq) return;
    hbf[i] = 0;
    c[i] = 0.0f;
}

// ---------------- fused LSTM step: gates GEMM (WMMA) + cell update ----------------
// Block = 256 thr (8 waves) owns 64 permuted gate columns = 16 h-units, all 32 batch rows.
// Grid = 32 blocks covers N'=2048. Gates live only in LDS.
__global__ void lstm_step(const unsigned short* __restrict__ hbf,
                          const unsigned short* __restrict__ xbf,
                          const unsigned short* __restrict__ WT,
                          const float* __restrict__ b,
                          float* __restrict__ cst,
                          unsigned short* __restrict__ hbf_out, // == hbf (in-place state)
                          float* __restrict__ lout,
                          int layer, int t) {
    __shared__ float gs[Bq][64];        // 32 rows x 64 permuted gate cols (8 KB)
    const int NB = blockIdx.x;          // 0..31

    int wave = threadIdx.x >> 5, lane = threadIdx.x & 31;
    int wm = wave & 1, wn = wave >> 1;  // 2 x 4 tile grid of 16x16
    int half = lane >> 4, r = lane & 15;
    int m0 = wm * 16;
    int nloc = wn * 16;                 // column offset inside block
    int row  = m0 + r;                  // batch row
    int ncol = NB * 64 + nloc + r;      // permuted global gate column

    const unsigned short* hrow = hbf + ((long)layer * Bq + row) * Hq;
    const unsigned short* xrow = (layer == 0)
        ? xbf + ((long)row * Tq + t) * Eq
        : hbf + ((long)(layer - 1) * Bq + row) * Hq;
    const unsigned short* bcol = WT + ((long)layer * (4 * Hq) + ncol) * (Hq + Eq);

    v8f acc = {0.f, 0.f, 0.f, 0.f, 0.f, 0.f, 0.f, 0.f};
    // recurrent half: K = 0..H over h_prev (loop-invariant base pointer)
    #pragma unroll 4
    for (int kk = 0; kk < Hq; kk += 32) {
        v16bf a  = load_a_frag(hrow, kk, half);
        v16bf bb = load_b_frag(bcol, kk, half);
        acc = __builtin_amdgcn_wmma_f32_16x16x32_bf16(
                  false, a, false, bb, (short)0, acc, false, false);
    }
    // input half: K = H..H+E over x / lower-layer h (loop-invariant base pointer)
    #pragma unroll 4
    for (int kk = 0; kk < Eq; kk += 32) {
        v16bf a  = load_a_frag(xrow, kk, half);
        v16bf bb = load_b_frag(bcol, Hq + kk, half);
        acc = __builtin_amdgcn_wmma_f32_16x16x32_bf16(
                  false, a, false, bb, (short)0, acc, false, false);
    }

    // bias: permuted col n' -> original col (n'&3)*H + (n'>>2)
    float bias = b[(long)layer * (4 * Hq) + (ncol & 3) * Hq + (ncol >> 2)];
    #pragma unroll
    for (int j = 0; j < 8; ++j)
        gs[m0 + half * 8 + j][nloc + r] = acc[j] + bias;
    __syncthreads();

    // cell update: 32 b x 16 h = 512 elements, 2 per thread
    #pragma unroll
    for (int e = threadIdx.x; e < Bq * 16; e += 256) {
        int bi = e >> 4, hl = e & 15;
        int h  = NB * 16 + hl;
        float f  = sigmoidf_(gs[bi][4 * hl + 0]);
        float ig = sigmoidf_(gs[bi][4 * hl + 1]);
        float gg = tanhf_  (gs[bi][4 * hl + 2]);
        float o  = sigmoidf_(gs[bi][4 * hl + 3]);
        long ci = ((long)layer * Bq + bi) * Hq + h;
        float cn = f * cst[ci] + ig * gg;
        cst[ci] = cn;
        float hn = o * tanhf_(cn);
        hbf_out[ci] = f32_to_bf16(hn);
        if (layer == Lq - 1)
            lout[((long)bi * Tq + t) * Hq + h] = hn;
    }
}

// ---------------- LayerNorm over rows of [B*T, H], emit bf16 A ----------------
__global__ void layernorm_rows(const float* __restrict__ out,
                               const float* __restrict__ gamma,
                               const float* __restrict__ beta,
                               unsigned short* __restrict__ A) {
    __shared__ float red[256];
    int tid = threadIdx.x;
    const float* x = out + (long)blockIdx.x * Hq;

    float s = 0.f;
    for (int j = tid; j < Hq; j += 256) s += x[j];
    red[tid] = s; __syncthreads();
    for (int st = 128; st > 0; st >>= 1) {
        if (tid < st) red[tid] += red[tid + st];
        __syncthreads();
    }
    float mu = red[0] * (1.0f / Hq);
    __syncthreads();

    float v = 0.f;
    for (int j = tid; j < Hq; j += 256) { float d = x[j] - mu; v += d * d; }
    red[tid] = v; __syncthreads();
    for (int st = 128; st > 0; st >>= 1) {
        if (tid < st) red[tid] += red[tid + st];
        __syncthreads();
    }
    float inv = rsqrtf(red[0] * (1.0f / Hq) + EPSq);

    unsigned short* arow = A + (long)blockIdx.x * Hq;
    for (int j = tid; j < Hq; j += 256)
        arow[j] = f32_to_bf16((x[j] - mu) * inv * gamma[j] + beta[j]);
}

// ---------------- Projection GEMM: [4096,512]bf16 @ [512,32000]bf16 + bh -> fp32 ----------------
// Block 256 thr (8 waves), tile 32(M) x 256(N). B tile staged in LDS via async
// global->LDS b128 copies, double-buffered on ASYNCcnt. LDS col stride 40 shorts
// (80 B): 16 B aligned and bank-conflict-free (20-bank stride pattern).
#define BSTRIDE 40
__global__ void proj_gemm(const unsigned short* __restrict__ A,
                          const unsigned short* __restrict__ BT,   // [V, H] bf16
                          const float* __restrict__ bh,
                          float* __restrict__ C) {
    __shared__ unsigned short Bs[2][256 * BSTRIDE];   // 2 x 20 KB

    int tid  = threadIdx.x;
    int wave = tid >> 5, lane = tid & 31;
    int wm = wave & 1, wn = wave >> 1;
    int half = lane >> 4, r = lane & 15;
    long m0     = (long)blockIdx.y * 32 + wm * 16;
    long n0blk  = (long)blockIdx.x * 256;

    const unsigned short* arow = A + (m0 + r) * Hq;
    const unsigned short* bblk = BT + n0blk * Hq;

    // each thread issues 4 async b128 copies per stage (16 KB total per stage)
    auto issue = [&](int stage, int kk) {
        #pragma unroll
        for (int seg = 0; seg < 4; ++seg) {
            int f = tid + seg * 256;          // 0..1023
            int col = f >> 2, part = f & 3;   // col 0..255, 16B chunk 0..3
            async_copy_b128(bblk + (long)col * Hq + kk + part * 8,
                            &Bs[stage][col * BSTRIDE + part * 8]);
        }
    };

    v8f acc0 = {0.f,0.f,0.f,0.f,0.f,0.f,0.f,0.f};
    v8f acc1 = acc0, acc2 = acc0, acc3 = acc0;

    issue(0, 0);
    int kk = 0;
    for (; kk < Hq - 32; kk += 32) {
        int cur = (kk >> 5) & 1;
        issue(cur ^ 1, kk + 32);
        WAIT_ASYNC(4);                 // oldest 4 (current stage) landed
        __syncthreads();

        __builtin_prefetch(arow + kk + 32, 0, 1);   // global_prefetch_b8 on A
        v16bf a = load_a_frag(arow, kk, half);
        const unsigned short* base = &Bs[cur][0];
        v16bf b0 = load_b_frag(base + (wn * 64 +  0 + r) * BSTRIDE, 0, half);
        v16bf b1 = load_b_frag(base + (wn * 64 + 16 + r) * BSTRIDE, 0, half);
        v16bf b2 = load_b_frag(base + (wn * 64 + 32 + r) * BSTRIDE, 0, half);
        v16bf b3 = load_b_frag(base + (wn * 64 + 48 + r) * BSTRIDE, 0, half);
        acc0 = __builtin_amdgcn_wmma_f32_16x16x32_bf16(false, a, false, b0, (short)0, acc0, false, false);
        acc1 = __builtin_amdgcn_wmma_f32_16x16x32_bf16(false, a, false, b1, (short)0, acc1, false, false);
        acc2 = __builtin_amdgcn_wmma_f32_16x16x32_bf16(false, a, false, b2, (short)0, acc2, false, false);
        acc3 = __builtin_amdgcn_wmma_f32_16x16x32_bf16(false, a, false, b3, (short)0, acc3, false, false);
        __syncthreads();               // stage reuse guard for next issue
    }
    {
        int cur = (kk >> 5) & 1;
        WAIT_ASYNC(0);
        __syncthreads();
        v16bf a = load_a_frag(arow, kk, half);
        const unsigned short* base = &Bs[cur][0];
        v16bf b0 = load_b_frag(base + (wn * 64 +  0 + r) * BSTRIDE, 0, half);
        v16bf b1 = load_b_frag(base + (wn * 64 + 16 + r) * BSTRIDE, 0, half);
        v16bf b2 = load_b_frag(base + (wn * 64 + 32 + r) * BSTRIDE, 0, half);
        v16bf b3 = load_b_frag(base + (wn * 64 + 48 + r) * BSTRIDE, 0, half);
        acc0 = __builtin_amdgcn_wmma_f32_16x16x32_bf16(false, a, false, b0, (short)0, acc0, false, false);
        acc1 = __builtin_amdgcn_wmma_f32_16x16x32_bf16(false, a, false, b1, (short)0, acc1, false, false);
        acc2 = __builtin_amdgcn_wmma_f32_16x16x32_bf16(false, a, false, b2, (short)0, acc2, false, false);
        acc3 = __builtin_amdgcn_wmma_f32_16x16x32_bf16(false, a, false, b3, (short)0, acc3, false, false);
    }

    const long N = Vq;
    v8f accs[4] = {acc0, acc1, acc2, acc3};
    #pragma unroll
    for (int j = 0; j < 4; ++j) {
        long nc = n0blk + wn * 64 + 16 * j + r;
        float bias = bh[nc];
        #pragma unroll
        for (int q = 0; q < 8; ++q) {
            long m = m0 + half * 8 + q;
            C[m * N + nc] = accs[j][q] + bias;
        }
    }
}

// ---------------- host launch ----------------
extern "C" void kernel_launch(void* const* d_in, const int* in_sizes, int n_in,
                              void* d_out, int out_size, void* d_ws, size_t ws_size,
                              hipStream_t stream) {
    const int*   idx   = (const int*)  d_in[0];
    const float* emb   = (const float*)d_in[1];
    const float* W     = (const float*)d_in[2];
    const float* b     = (const float*)d_in[3];
    const float* gamma = (const float*)d_in[4];
    const float* beta  = (const float*)d_in[5];
    const float* Wh    = (const float*)d_in[6];
    const float* bh    = (const float*)d_in[7];
    float* out_logits  = (float*)d_out;

    char* ws = (char*)d_ws;
    size_t off = 0;
    auto carve = [&](size_t bytes) -> void* {
        void* p = ws + off;
        off += (bytes + 255) & ~(size_t)255;
        return p;
    };
    unsigned short* WT   = (unsigned short*)carve((size_t)Lq * 4 * Hq * (Hq + Eq) * 2);
    unsigned short* WhT  = (unsigned short*)carve((size_t)Vq * Hq * 2);
    unsigned short* xbf  = (unsigned short*)carve((size_t)Bq * Tq * Eq * 2);
    unsigned short* hbf  = (unsigned short*)carve((size_t)Lq * Bq * Hq * 2);
    float*          cst  = (float*)carve((size_t)Lq * Bq * Hq * 4);
    float*          lout = (float*)carve((size_t)Bq * Tq * Hq * 4);
    unsigned short* Abf  = (unsigned short*)carve((size_t)Bq * Tq * Hq * 2);
    (void)ws_size;

    // prep: weight conversion/transpose (gate-interleaved), embed gather, state init
    {
        long tot = (long)Lq * 4 * Hq * (Hq + Eq);
        prep_w<<<(unsigned)((tot + 255) / 256), 256, 0, stream>>>(W, WT);
    }
    {
        long tot = (long)Vq * Hq;
        prep_wh<<<(unsigned)((tot + 255) / 256), 256, 0, stream>>>(Wh, WhT);
    }
    {
        long tot = (long)Bq * Tq * Eq;
        embed_gather<<<(unsigned)((tot + 255) / 256), 256, 0, stream>>>(idx, emb, xbf);
    }
    init_state<<<(Lq * Bq * Hq + 255) / 256, 256, 0, stream>>>(hbf, cst);

    // sequential LSTM: one fused kernel per (t, layer); stream order carries recurrence
    for (int t = 0; t < Tq; ++t)
        for (int layer = 0; layer < Lq; ++layer)
            lstm_step<<<32, 256, 0, stream>>>(hbf, xbf, WT, b, cst, hbf, lout, layer, t);

    // LayerNorm per row
    layernorm_rows<<<Bq * Tq, 256, 0, stream>>>(lout, gamma, beta, Abf);

    // output projection (async-LDS staged, double-buffered)
    dim3 pg(Vq / 256, (Bq * Tq) / 32);   // (125, 128)
    proj_gemm<<<pg, 256, 0, stream>>>(Abf, WhT, bh, out_logits);
}
